// QuantLinear_25091198943821
// MI455X (gfx1250) — matrix operand order repaired and checked by reading, not compiled, not run on hardware
//
#include <hip/hip_runtime.h>

// W4A32 quantized linear as an f16-WMMA GEMM for gfx1250 (MI455X).
// M=8192 (B*S), K=4096 (IN), N=11008 (OUT), groupsize=128, bits=4.
// Compute-bound at WMMA-f16 rate (AI ~1400 FLOP/byte >> machine balance).

typedef __attribute__((ext_vector_type(16))) _Float16 v16h;
typedef __attribute__((ext_vector_type(8)))  _Float16 v8h;
typedef __attribute__((ext_vector_type(4)))  _Float16 v4h;
typedef __attribute__((ext_vector_type(2)))  _Float16 v2h;
typedef __attribute__((ext_vector_type(8)))  float    v8f;
typedef __attribute__((ext_vector_type(4)))  float    v4f;

union FragAB { v16h v; struct { v8h lo, hi; } p; };
union BPack  { v8h v; unsigned u[4]; };

#define TM 128
#define TN 128
#define TK 64     // two WMMA k-steps (32 each) per LDS stage
#define LDA 72    // padded LDS stride (halves): 144B = 9*16B, conflict-spreading
#define LDB 72
#define THREADS 256

__global__ __launch_bounds__(THREADS) void qgemm_w4a32(
    const float* __restrict__ X,   // (M, K) f32
    const int*   __restrict__ QW,  // (K/8, N) int32, 8 x 4-bit along K
    const float* __restrict__ SC,  // (G, N) f32
    const int*   __restrict__ QZ,  // (G, N/8) int32, 8 x 4-bit along N
    float* __restrict__ C,         // (M, N) f32
    int M, int N, int K) {
  __shared__ __align__(16) _Float16 As[TM * LDA];  // [m][k] row-major
  __shared__ __align__(16) _Float16 Bs[TN * LDB];  // [n][k] N-major (k contiguous)

  const int tid  = threadIdx.x;
  const int lane = tid & 31;
  const int l15  = lane & 15;
  const int wave = tid >> 5;
  const int wm   = wave >> 2;  // 0..1 : 64 rows per wave
  const int wn   = wave & 3;   // 0..3 : 32 cols per wave
  const int m0 = blockIdx.y * TM;
  const int n0 = blockIdx.x * TN;

  v8f acc[4][2] = {};  // 4 M-tiles x 2 N-tiles of 16x16 f32

  // ---- loop-invariant per-thread slots ----
  // A staging: 128x64 f32 tile, 16 threads per row (a float4 each), 8 row-strips.
  const int aseg = tid & 15;
  const int arow = tid >> 4;
  const float* xp[8];
  _Float16*    asp[8];
  #pragma unroll
  for (int r = 0; r < 8; ++r) {
    const int rr = arow + 16 * r;
    xp[r]  = X + (size_t)(m0 + rr) * K + aseg * 4;
    asp[r] = As + rr * LDA + aseg * 4;
  }
  // B dequant: 64K x 128N tile = 1024 packed int32s, 4 per thread.
  int dq_gn[4];
  const int* qp[4];
  _Float16*  bsp[4];
  #pragma unroll
  for (int t = 0; t < 4; ++t) {
    const int idx = tid + t * THREADS;  // (8 kp) x (128 n)
    const int kp  = idx >> 7;
    const int n   = idx & 127;
    dq_gn[t] = n0 + n;
    qp[t]  = QW + (size_t)kp * N + (n0 + n);
    bsp[t] = Bs + n * LDB + kp * 8;
  }

  const int nGroups = K >> 7;  // 32 groups of 128 K (2 LDS tiles each)

  #pragma unroll 1
  for (int g = 0; g < nGroups; ++g) {
    // ---- per-group dequant coefficients: w = (1024+q)*s + (-(1024+z)*s) ----
    v2h s2[4], b2[4];
    #pragma unroll
    for (int t = 0; t < 4; ++t) {
      const int gn = dq_gn[t];
      const float    s  = SC[(size_t)g * N + gn];
      const unsigned zq = (unsigned)QZ[(size_t)g * (N >> 3) + (gn >> 3)];
      const float    z  = (float)((zq >> ((gn & 7) * 4)) & 0xF);
      const _Float16 sh = (_Float16)s;
      const _Float16 bh = (_Float16)(-(1024.0f + z) * s);
      s2[t] = (v2h){sh, sh};
      b2[t] = (v2h){bh, bh};
    }

    #pragma unroll
    for (int half = 0; half < 2; ++half) {
      const int k0 = (g * 2 + half) * TK;

      // ---- stage A tile into registers: f32 -> f16 ----
      v4h areg[8];
      #pragma unroll
      for (int r = 0; r < 8; ++r) {
        v4f xv = *(const v4f*)(xp[r] + k0);
        areg[r] = __builtin_convertvector(xv, v4h);
      }
      // Speculative prefetch of the next K-slab (dropped silently if OOB).
      __builtin_prefetch(xp[0] + k0 + TK, 0, 3);

      // ---- stage B tile: fast 4-bit dequant via 0x6400 bias + pk_fma ----
      const size_t qrow = (size_t)(g * 16 + half * 8) * N;
      v8h breg[4];
      #pragma unroll
      for (int t = 0; t < 4; ++t) {
        const unsigned q = (unsigned)qp[t][qrow];
        BPack pk;
        #pragma unroll
        for (int p = 0; p < 4; ++p) {
          // pair (nibble 2p, nibble 2p+1) -> packed f16x2 of (1024+q)
          const unsigned w = ((q >> (8 * p)) & 0xFu) |
                             (((q >> (8 * p + 4)) & 0xFu) << 16) | 0x64006400u;
          v2h wv = __builtin_bit_cast(v2h, w);
          wv = wv * s2[t] + b2[t];  // v_pk_fma_f16
          pk.u[p] = __builtin_bit_cast(unsigned, wv);
        }
        breg[t] = pk.v;
      }

      __syncthreads();  // previous iteration's fragment reads done

      #pragma unroll
      for (int r = 0; r < 8; ++r)
        *(v4h*)__builtin_assume_aligned(asp[r], 8) = areg[r];
      #pragma unroll
      for (int t = 0; t < 4; ++t)
        *(v8h*)__builtin_assume_aligned(bsp[t], 16) = breg[t];

      __syncthreads();

      // ---- two WMMA k-steps per staged tile ----
      #pragma unroll
      for (int s = 0; s < 2; ++s) {
        const int ks = s * 32;

        // B 32x16 f16: lanes 0-15 hold K=0..15 of column N=lane; lanes 16-31 K=16..31.
        FragAB b[2];
        const int kbB = ks + ((lane < 16) ? 0 : 16);
        #pragma unroll
        for (int ni = 0; ni < 2; ++ni) {
          const _Float16* p = Bs + (wn * 32 + ni * 16 + l15) * LDB + kbB;
          b[ni].p.lo = *(const v8h*)__builtin_assume_aligned(p, 16);
          b[ni].p.hi = *(const v8h*)__builtin_assume_aligned(p + 8, 16);
        }
        // A 16x32 f16: lanes 0-15 row M=lane hold K=0..15;
        // lanes 16-31 hold K=16..31.
        FragAB a[4];
        const int kbA = ks + ((lane < 16) ? 0 : 16);
        #pragma unroll
        for (int mi = 0; mi < 4; ++mi) {
          const _Float16* p = As + (wm * 64 + mi * 16 + l15) * LDA + kbA;
          a[mi].p.lo = *(const v8h*)__builtin_assume_aligned(p, 16);
          a[mi].p.hi = *(const v8h*)__builtin_assume_aligned(p + 8, 16);
        }
        #pragma unroll
        for (int mi = 0; mi < 4; ++mi)
          #pragma unroll
          for (int ni = 0; ni < 2; ++ni)
            acc[mi][ni] = __builtin_amdgcn_wmma_f32_16x16x32_f16(
                /*neg_a=*/false, a[mi].v, /*neg_b=*/false, b[ni].v,
                /*c_mod=*/(short)0, acc[mi][ni],
                /*reuse_a=*/false, /*reuse_b=*/false);
      }
    }
  }

  // ---- epilogue: C/D f32 layout: VGPR r -> M=r (lanes 0-15) / M=r+8 (16-31), N=lane&15
  const int rowoff = (lane >> 4) << 3;
  #pragma unroll
  for (int mi = 0; mi < 4; ++mi) {
    #pragma unroll
    for (int ni = 0; ni < 2; ++ni) {
      const int gm = m0 + wm * 64 + mi * 16 + rowoff;
      const int gn = n0 + wn * 32 + ni * 16 + l15;
      float* p = C + (size_t)gm * N + gn;
      #pragma unroll
      for (int r = 0; r < 8; ++r) {
        *p = acc[mi][ni][r];
        p += N;
      }
    }
  }
}

extern "C" void kernel_launch(void* const* d_in, const int* in_sizes, int n_in,
                              void* d_out, int out_size, void* d_ws, size_t ws_size,
                              hipStream_t stream) {
  const float* X  = (const float*)d_in[0];  // (B,S,IN) f32
  const int*   QW = (const int*)d_in[1];    // (IN/8, OUT) int32
  const float* SC = (const float*)d_in[2];  // (G, OUT) f32
  const int*   QZ = (const int*)d_in[3];    // (G, OUT/8) int32
  float* OUT = (float*)d_out;

  // Shapes fixed by the reference setup: IN=4096, OUT=11008, M=B*S.
  const int K = 4096;
  const int M = in_sizes[0] / K;            // 8192
  const int N = in_sizes[1] / (K / 8);      // 11008

  dim3 grid(N / TN, M / TM);                // 86 x 64 workgroups
  qgemm_w4a32<<<grid, dim3(THREADS), 0, stream>>>(X, QW, SC, QZ, OUT, M, N, K);
}
